// OurNet_26474178413219
// MI455X (gfx1250) — matrix operand order repaired and checked by reading, not compiled, not run on hardware
//
#include <hip/hip_runtime.h>

// ---------------------------------------------------------------------------
// 2-layer LSTM (T=30, B=16384, H=100, P=7) on gfx1250 WMMA.
// Layer-1 input GEMM collapsed to rank-1 outer product (xi = x*W1 + b1).
// Layer-2 input GEMM hoisted out of the 7-step loop (input is constant).
// Recurrent h @ Whh^T done as v_wmma_f32_16x16x32_f16 with h staged in LDS.
// ---------------------------------------------------------------------------

typedef __attribute__((ext_vector_type(16))) _Float16 v16h;
typedef __attribute__((ext_vector_type(8)))  _Float16 v8h;
typedef __attribute__((ext_vector_type(8)))  float    v8f;

#define T_    30
#define B_    16384
#define H_    100
#define P_    7
#define G4    400      // 4*H
#define HP    128      // H padded to k-chunk multiple (4 chunks of 32)
#define BT    32       // batch rows per workgroup (2 WMMA M-tiles)
#define GSTR  404      // gate-buffer row stride (floats), padded vs 400 to dodge bank conflicts
#define NTILE 25       // 400 / 16 N-tiles

__device__ __forceinline__ float sigm(float x)  { return 1.f / (1.f + __expf(-x)); }
__device__ __forceinline__ float ftanh(float x) { float e = __expf(2.f * x); return 1.f - 2.f / (e + 1.f); }

// gates[0:32][0:400] = hbuf(f16, [32][128]) @ wB(f16, [400][128])^T   (K = 128, zero-padded)
// wB is row-major [n][k] in global (L2-resident), exactly B[k][n] = W[n][k].
// `wavu` must be wave-uniform *scalar* (readfirstlane'd) so the nt loop uses
// scalar branching and EXEC provably stays all-1s around every WMMA.
__device__ __forceinline__ void gemm_step(const _Float16* __restrict__ hb,   // LDS  [BT][HP]
                                          const _Float16* __restrict__ wB,   // glob [G4][HP]
                                          float* __restrict__ outg,          // LDS  [BT][GSTR]
                                          int lane, int wavu)
{
  const int rn = lane & 15;   // N column within tile (B/C/D layout) and M row (A layout)
  const int hi = lane >> 4;   // half-wave selector

  // A tiles are invariant across nt: load once (compiler hoists these ds_load_b128s).
  v16h A0[4], A1[4];
#pragma unroll
  for (int kc = 0; kc < 4; ++kc) {
    const _Float16* ap0 = hb + rn * HP        + hi * 8 + kc * 32;
    const _Float16* ap1 = hb + (16 + rn) * HP + hi * 8 + kc * 32;
    v8h a0lo = *(const v8h*)(ap0);
    v8h a0hi = *(const v8h*)(ap0 + 16);
    v8h a1lo = *(const v8h*)(ap1);
    v8h a1hi = *(const v8h*)(ap1 + 16);
#pragma unroll
    for (int i = 0; i < 8; ++i) {
      A0[kc][i] = a0lo[i]; A0[kc][i + 8] = a0hi[i];
      A1[kc][i] = a1lo[i]; A1[kc][i + 8] = a1hi[i];
    }
  }

  for (int nt = wavu; nt < NTILE; nt += 8) {
    v8f acc0 = {}; v8f acc1 = {};
    const int ncol = nt * 16 + rn;
    const _Float16* bp = wB + ncol * HP + hi * 8;      // B: lane holds K = kc*32 + hi*8 + e (e<8), kc*32 + 16 + hi*8 + (e-8) (e>=8)
#pragma unroll
    for (int kc = 0; kc < 4; ++kc) {
      v8h blo = *(const v8h*)(bp + kc * 32);
      v8h bhi = *(const v8h*)(bp + kc * 32 + 16);
      v16h Bv = __builtin_shufflevector(blo, bhi, 0, 1, 2, 3, 4, 5, 6, 7, 8, 9, 10, 11, 12, 13, 14, 15);
      acc0 = __builtin_amdgcn_wmma_f32_16x16x32_f16(false, A0[kc], false, Bv, (short)0, acc0, false, false);
      acc1 = __builtin_amdgcn_wmma_f32_16x16x32_f16(false, A1[kc], false, Bv, (short)0, acc1, false, false);
    }
    // C/D layout: lane l, vgpr r -> M = (l/16)*8 + r, N = l%16
    const int mr = hi * 8;
#pragma unroll
    for (int r = 0; r < 8; ++r) {
      outg[(mr + r) * GSTR + ncol]      = acc0[r];
      outg[(16 + mr + r) * GSTR + ncol] = acc1[r];
    }
  }
}

// ---------------------------------------------------------------------------
// Prep: f32 -> f16 weight repack (K padded to 128) + rank-1 constants:
//   u1[n]  = (Wih1 @ W1)[n],  v1c[n] = (Wih1 @ b1)[n] + bih1[n] + bhh1[n]
//   bias2[n] = bih2[n] + bhh2[n]
// ---------------------------------------------------------------------------
__global__ __launch_bounds__(128)
void lstm_prep(const float* __restrict__ W1,   const float* __restrict__ b1,
               const float* __restrict__ Wih1, const float* __restrict__ Whh1,
               const float* __restrict__ bih1, const float* __restrict__ bhh1,
               const float* __restrict__ Wih2, const float* __restrict__ Whh2,
               const float* __restrict__ bih2, const float* __restrict__ bhh2,
               _Float16* __restrict__ whh1h, _Float16* __restrict__ whh2h,
               _Float16* __restrict__ wih2h,
               float* __restrict__ u1, float* __restrict__ v1c, float* __restrict__ bias2)
{
  __shared__ float su[128], sv[128];
  const int n = blockIdx.x;     // 0..399
  const int k = threadIdx.x;    // 0..127
  const bool in = (k < H_);
  whh1h[n * HP + k] = (_Float16)(in ? Whh1[n * H_ + k] : 0.f);
  whh2h[n * HP + k] = (_Float16)(in ? Whh2[n * H_ + k] : 0.f);
  wih2h[n * HP + k] = (_Float16)(in ? Wih2[n * H_ + k] : 0.f);
  const float wi1 = in ? Wih1[n * H_ + k] : 0.f;
  su[k] = wi1 * (in ? W1[k] : 0.f);
  sv[k] = wi1 * (in ? b1[k] : 0.f);
  __syncthreads();
  for (int s = 64; s > 0; s >>= 1) {
    if (k < s) { su[k] += su[k + s]; sv[k] += sv[k + s]; }
    __syncthreads();
  }
  if (k == 0) {
    u1[n]    = su[0];
    v1c[n]   = sv[0] + bih1[n] + bhh1[n];
    bias2[n] = bih2[n] + bhh2[n];
  }
}

// ---------------------------------------------------------------------------
// Main: persistent batch tile of 32 rows per workgroup, 8 wave32.
// ---------------------------------------------------------------------------
__global__ __launch_bounds__(256)
void lstm_main(const float* __restrict__ x,
               const _Float16* __restrict__ whh1h,
               const _Float16* __restrict__ whh2h,
               const _Float16* __restrict__ wih2h,
               const float* __restrict__ u1g,
               const float* __restrict__ v1g,
               const float* __restrict__ b2g,
               const float* __restrict__ W2,
               const float* __restrict__ b2,
               float* __restrict__ out)
{
  extern __shared__ char smem[];
  _Float16* hbuf = (_Float16*)smem;                                   // [BT][HP]   f16
  float*    cbuf = (float*)(smem + BT * HP * 2);                      // [BT][H]    f32
  float*    gates = (float*)(smem + BT * HP * 2 + BT * H_ * 4);       // [BT][GSTR] f32
  float*    z2   = gates + BT * GSTR;                                 // [BT][GSTR] f32
  float*    su   = z2 + BT * GSTR;                                    // [G4]
  float*    sv   = su + G4;
  float*    sb2  = sv + G4;
  float*    sx   = sb2 + G4;                                          // [BT] staged x column

  const int tid  = threadIdx.x;
  const int lane = tid & 31;
  const int wavu = __builtin_amdgcn_readfirstlane(tid >> 5);  // scalar wave id
  const int b0   = blockIdx.x * BT;

  for (int i = tid; i < BT * HP; i += 256) hbuf[i] = (_Float16)0.f;
  for (int i = tid; i < BT * H_; i += 256) cbuf[i] = 0.f;
  for (int i = tid; i < G4; i += 256) { su[i] = u1g[i]; sv[i] = v1g[i]; sb2[i] = b2g[i]; }
  __syncthreads();

  float yacc = 0.f;

  // ---------------- layer 1: 30 sequential steps ----------------
  for (int t = 0; t < T_; ++t) {
    if (tid < BT) sx[tid] = x[t * B_ + b0 + tid];   // stage x column once per step
    gemm_step(hbuf, whh1h, gates, lane, wavu);
    __syncthreads();                                // covers gates AND sx
    for (int idx = tid; idx < BT * H_; idx += 256) {
      const int b = idx / H_, hid = idx - b * H_;
      const float xv = sx[b];
      const float* g = gates + b * GSTR;
      const float gi = g[hid]           + xv * su[hid]           + sv[hid];
      const float gf = g[H_ + hid]      + xv * su[H_ + hid]      + sv[H_ + hid];
      const float gg = g[2 * H_ + hid]  + xv * su[2 * H_ + hid]  + sv[2 * H_ + hid];
      const float go = g[3 * H_ + hid]  + xv * su[3 * H_ + hid]  + sv[3 * H_ + hid];
      const float c = sigm(gf) * cbuf[idx] + sigm(gi) * ftanh(gg);
      const float h = sigm(go) * ftanh(c);
      cbuf[idx] = c;
      hbuf[b * HP + hid] = (_Float16)h;
    }
    __syncthreads();
  }

  // z2 = last @ Wih2^T  (computed once; input to layer-2 is constant)
  gemm_step(hbuf, wih2h, z2, lane, wavu);
  __syncthreads();

  // ---------------- layer 2: 7 sequential steps + fused FC2 ----------------
  for (int p = 0; p < P_; ++p) {
    gemm_step(hbuf, whh2h, gates, lane, wavu);
    __syncthreads();
    for (int idx = tid; idx < BT * H_; idx += 256) {
      const int b = idx / H_, hid = idx - b * H_;
      const float* g = gates + b * GSTR;
      const float* z = z2 + b * GSTR;
      const float gi = g[hid]          + z[hid]          + sb2[hid];
      const float gf = g[H_ + hid]     + z[H_ + hid]     + sb2[H_ + hid];
      const float gg = g[2 * H_ + hid] + z[2 * H_ + hid] + sb2[2 * H_ + hid];
      const float go = g[3 * H_ + hid] + z[3 * H_ + hid] + sb2[3 * H_ + hid];
      const float c = sigm(gf) * cbuf[idx] + sigm(gi) * ftanh(gg);
      const float h = sigm(go) * ftanh(c);
      cbuf[idx] = c;
      hbuf[b * HP + hid] = (_Float16)h;
    }
    __syncthreads();
    if (tid < BT * P_) {               // 224 threads: one (batch row, output col) each
      const int b = tid / P_, po = tid - b * P_;
      const float* w = W2 + po * (P_ * H_) + p * H_;
      const _Float16* hr = hbuf + b * HP;
      float s = 0.f;
#pragma unroll 4
      for (int hid = 0; hid < H_; ++hid) s = fmaf((float)hr[hid], w[hid], s);
      yacc += s;
    }
    __syncthreads();
  }

  if (tid < BT * P_) {
    const int b = tid / P_, po = tid - b * P_;
    out[(b0 + b) * P_ + po] = sigm(yacc + b2[po]);
  }
}

// ---------------------------------------------------------------------------
extern "C" void kernel_launch(void* const* d_in, const int* in_sizes, int n_in,
                              void* d_out, int out_size, void* d_ws, size_t ws_size,
                              hipStream_t stream)
{
  const float* x    = (const float*)d_in[0];
  const float* W1   = (const float*)d_in[1];
  const float* b1   = (const float*)d_in[2];
  const float* Wih1 = (const float*)d_in[3];
  const float* Whh1 = (const float*)d_in[4];
  const float* bih1 = (const float*)d_in[5];
  const float* bhh1 = (const float*)d_in[6];
  const float* Wih2 = (const float*)d_in[7];
  const float* Whh2 = (const float*)d_in[8];
  const float* bih2 = (const float*)d_in[9];
  const float* bhh2 = (const float*)d_in[10];
  const float* W2   = (const float*)d_in[11];
  const float* b2   = (const float*)d_in[12];
  float* out = (float*)d_out;

  // workspace layout (bytes):
  //   0      : whh1h  f16 [400][128]  (102400)
  //   102400 : whh2h  f16 [400][128]  (102400)
  //   204800 : wih2h  f16 [400][128]  (102400)
  //   307200 : u1     f32 [400]
  //   308800 : v1c    f32 [400]
  //   310400 : bias2  f32 [400]
  char* ws = (char*)d_ws;
  _Float16* whh1h = (_Float16*)(ws);
  _Float16* whh2h = (_Float16*)(ws + 102400);
  _Float16* wih2h = (_Float16*)(ws + 204800);
  float* u1    = (float*)(ws + 307200);
  float* v1c   = (float*)(ws + 308800);
  float* bias2 = (float*)(ws + 310400);

  lstm_prep<<<G4, 128, 0, stream>>>(W1, b1, Wih1, Whh1, bih1, bhh1,
                                    Wih2, Whh2, bih2, bhh2,
                                    whh1h, whh2h, wih2h, u1, v1c, bias2);

  const size_t smem = (size_t)BT * HP * 2      // hbuf
                    + (size_t)BT * H_ * 4      // cbuf
                    + 2u * BT * GSTR * 4       // gates + z2
                    + 3u * G4 * 4              // su, sv, sb2
                    + (size_t)BT * 4;          // sx
  lstm_main<<<B_ / BT, 256, smem, stream>>>(x, whh1h, whh2h, wih2h,
                                            u1, v1c, bias2, W2, b2, out);
}
